// XEyTransformerLayer_78434692759832
// MI455X (gfx1250) — compile-verified, matrix operand order checked by reading, not run on hardware
//
#include <hip/hip_runtime.h>
#include <hip/hip_bf16.h>

// ---- problem constants (match reference) ----
#define BSc  4
#define NNc  256
#define DXc  128
#define DEc  64
#define DYc  128
#define Hc   8
#define DFc  16
#define FFXc 512
#define FFEc 128
#define FFYc 512

typedef __attribute__((ext_vector_type(16))) _Float16 v16h;
typedef __attribute__((ext_vector_type(8)))  float    v8f;

// ---------------------------------------------------------------------------
// k-index inside a 16x32 f16 A/B fragment, per ISA 7.12.2:
// lane l holds row (l&15); element j holds K = j + 8*((l>=16)+(j>=8)).
// ---------------------------------------------------------------------------
__device__ __forceinline__ int frag_k(int j, int lane) {
  return j + 8 * ((lane >> 4) + (j >> 3));
}

// ---------------------------------------------------------------------------
// One-time weight swizzle: fp32 W[Nout,K] -> f16 fragment-native layout.
// Fragment index fi = (ntile*(K/32) + ik)*32 + lane holds the 16 halves lane
// needs for that B tile/k-step; a B fragment load is then ONE aligned 32B read.
// ---------------------------------------------------------------------------
__global__ void swizzle_w(const float* __restrict__ W, _Float16* __restrict__ out,
                          int Nout, int K)
{
  int idx = blockIdx.x * blockDim.x + threadIdx.x;
  int total = Nout * K;
  if (idx >= total) return;
  int j    = idx & 15;
  int fi   = idx >> 4;
  int lane = fi & 31;
  int rest = fi >> 5;               // ntile*(K/32) + ik
  int nk   = K >> 5;
  int ntile = rest / nk;
  int ik    = rest - ntile * nk;
  int row = ntile * 16 + (lane & 15);
  int k   = (ik << 5) + frag_k(j, lane);
  out[idx] = (_Float16)W[(size_t)row * K + k];
}

// ---------------------------------------------------------------------------
// Async-stage a 16x32-float tile into LDS (128 x b128, 4 per lane, ASYNCcnt).
// ---------------------------------------------------------------------------
__device__ __forceinline__ void stage_a_tile(const float* __restrict__ A,
                                             int rowBase, int ldk, int kk,
                                             unsigned ldsBase, int lane) {
#pragma unroll
  for (int t = 0; t < 4; ++t) {
    int e = t * 32 + lane;
    int row = e >> 3, seg = e & 7;
    const float* gp = A + (size_t)(rowBase + row) * ldk + kk + seg * 4;
    unsigned lds = ldsBase + (unsigned)e * 16u;
    unsigned long long ga = (unsigned long long)(uintptr_t)gp;
    asm volatile("global_load_async_to_lds_b128 %0, %1, off"
                 :: "v"(lds), "v"(ga) : "memory");
  }
}

// ---------------------------------------------------------------------------
// Strip GEMM: out[M,Nout] = A[M,K] @ W[Nout,K]^T + bias, W pre-swizzled f16.
// One wave: 16 x (TN*16) strip; A staged via async LDS DMA (double buffered),
// fragment built once per k-step and reused across TN WMMAs.
// MASK: 0 none, 1 xm (row = b*N+q), 2 em (row = (b*N+q)*N+k).
// ---------------------------------------------------------------------------
template<int RELU, int MASK, int TN>
__global__ void gemm_wmma(const float* __restrict__ A, const _Float16* __restrict__ Wh,
                          const float* __restrict__ bias, float* __restrict__ out,
                          int M, int Kdim, int Nout, const int* __restrict__ nmask)
{
  __shared__ float smem[8][2][512];
  int w    = threadIdx.x >> 5;
  int lane = threadIdx.x & 31;
  int wave = (int)((blockIdx.x * blockDim.x + threadIdx.x) >> 5);
  int chunks = Nout / (16 * TN);
  int tile_m = wave / chunks;
  int chunk  = wave - tile_m * chunks;
  if (tile_m * 16 >= M) return;

  unsigned base0 = (unsigned)(uintptr_t)&smem[w][0][0];
  int rowBase = tile_m * 16;
  int nk = Kdim >> 5;
  const v16h* WB = (const v16h*)Wh;

  v8f acc[TN];
#pragma unroll
  for (int tn = 0; tn < TN; ++tn) acc[tn] = (v8f){};

  stage_a_tile(A, rowBase, Kdim, 0, base0, lane);
  for (int i = 0; i < nk; ++i) {
    if (i + 1 < nk) {
      stage_a_tile(A, rowBase, Kdim, (i + 1) << 5, base0 + ((i + 1) & 1) * 2048u, lane);
      asm volatile("s_wait_asynccnt 0x4" ::: "memory");
    } else {
      asm volatile("s_wait_asynccnt 0x0" ::: "memory");
    }
    int par = i & 1;
    v16h a;
    int ar = (lane & 15) * 32;
#pragma unroll
    for (int j = 0; j < 16; ++j) a[j] = (_Float16)smem[w][par][ar + frag_k(j, lane)];
#pragma unroll
    for (int tn = 0; tn < TN; ++tn) {
      v16h b = WB[(size_t)((chunk * TN + tn) * nk + i) * 32 + lane];
      acc[tn] = __builtin_amdgcn_wmma_f32_16x16x32_f16(false, a, false, b,
                                                       (short)0, acc[tn], false, false);
    }
  }

  float rmask[8];
#pragma unroll
  for (int j = 0; j < 8; ++j) {
    int row = rowBase + j + 8 * (lane >> 4);
    if (MASK == 1) {
      rmask[j] = (nmask[row] != 0) ? 1.f : 0.f;
    } else if (MASK == 2) {
      int b_ = row >> 16, rem = row & 65535;
      int q = rem >> 8, k2 = rem & 255;
      rmask[j] = ((nmask[b_ * NNc + q] != 0) && (nmask[b_ * NNc + k2] != 0)) ? 1.f : 0.f;
    } else {
      rmask[j] = 1.f;
    }
  }
#pragma unroll
  for (int tn = 0; tn < TN; ++tn) {
    int col = (chunk * TN + tn) * 16 + (lane & 15);
    float bcol = bias[col];
#pragma unroll
    for (int j = 0; j < 8; ++j) {
      int row = rowBase + j + 8 * (lane >> 4);
      float v = acc[tn][j] + bcol;
      if (RELU) v = v > 0.f ? v : 0.f;
      v *= rmask[j];
      out[(size_t)row * Nout + col] = v;
    }
  }
}

// ---------------------------------------------------------------------------
// Fused E kernel (weights pre-swizzled): one wave per 16 E-rows, all 8 heads.
// ---------------------------------------------------------------------------
__global__ void fused_e_attn(const float* __restrict__ E,
                             const _Float16* __restrict__ WaddH, const float* __restrict__ bAdd,
                             const _Float16* __restrict__ WmulH, const float* __restrict__ bMul,
                             const float* __restrict__ Qm, const float* __restrict__ Km,
                             const float* __restrict__ ye1, const float* __restrict__ ye2,
                             const int* __restrict__ nmask,
                             float* __restrict__ T, float* __restrict__ scores)
{
  __shared__ float sE[8][1024];
  const int R = BSc * NNc * NNc;
  int w    = threadIdx.x >> 5;
  int lane = threadIdx.x & 31;
  int tile_m = (int)((blockIdx.x * blockDim.x + threadIdx.x) >> 5);
  if (tile_m * 16 >= R) return;
  int rowBase = tile_m * 16;
  const v16h* WA = (const v16h*)WaddH;
  const v16h* WM = (const v16h*)WmulH;

  {
    unsigned base = (unsigned)(uintptr_t)&sE[w][0];
#pragma unroll
    for (int t = 0; t < 8; ++t) {
      int e = t * 32 + lane;
      int row = e >> 4, seg = e & 15;
      const float* gp = E + (size_t)(rowBase + row) * DEc + seg * 4;
      unsigned lds = base + (unsigned)e * 16u;
      unsigned long long ga = (unsigned long long)(uintptr_t)gp;
      asm volatile("global_load_async_to_lds_b128 %0, %1, off"
                   :: "v"(lds), "v"(ga) : "memory");
    }
    asm volatile("s_wait_asynccnt 0x0" ::: "memory");
  }

  int rj[8], bj[8], qj[8], kj[8];
  float emj[8];
#pragma unroll
  for (int j = 0; j < 8; ++j) {
    int r = rowBase + j + 8 * (lane >> 4);
    int b_ = r >> 16, rem = r & 65535;
    rj[j] = r; bj[j] = b_; qj[j] = rem >> 8; kj[j] = rem & 255;
    emj[j] = ((nmask[b_ * NNc + qj[j]] != 0) && (nmask[b_ * NNc + kj[j]] != 0)) ? 1.f : 0.f;
  }

  for (int g = 0; g < 2; ++g) {
    v8f c1[4], c2[4];
#pragma unroll
    for (int hh = 0; hh < 4; ++hh) { c1[hh] = (v8f){}; c2[hh] = (v8f){}; }

#pragma unroll
    for (int i = 0; i < 2; ++i) {            // K = 64 -> 2 k-steps
      int kk = i << 5;
      v16h a;
      int ar = (lane & 15) * DEc + kk;
#pragma unroll
      for (int j = 0; j < 16; ++j) a[j] = (_Float16)sE[w][ar + frag_k(j, lane)];
#pragma unroll
      for (int hh = 0; hh < 4; ++hh) {
        int head = g * 4 + hh;
        v16h b1 = WA[(size_t)(head * 2 + i) * 32 + lane];
        v16h b2 = WM[(size_t)(head * 2 + i) * 32 + lane];
        c1[hh] = __builtin_amdgcn_wmma_f32_16x16x32_f16(false, a, false, b1,
                                                        (short)0, c1[hh], false, false);
        c2[hh] = __builtin_amdgcn_wmma_f32_16x16x32_f16(false, a, false, b2,
                                                        (short)0, c2[hh], false, false);
      }
    }

#pragma unroll
    for (int hh = 0; hh < 4; ++hh) {
      int head = g * 4 + hh;
      int col = head * 16 + (lane & 15);
      float ba = bAdd[col], bm = bMul[col];
#pragma unroll
      for (int j = 0; j < 8; ++j) {
        float em = emj[j];
        float e1 = (c1[hh][j] + ba) * em;
        float e2 = (c2[hh][j] + bm) * em;
        float qk = Qm[(size_t)(bj[j] * NNc + qj[j]) * DXc + col] *
                   Km[(size_t)(bj[j] * NNc + kj[j]) * DXc + col] * 0.25f;
        float Av = e1 + (e2 + 1.f) * qk;

        float s = Av;
#pragma unroll
        for (int off = 1; off < 16; off <<= 1) s += __shfl_xor(s, off, 32);
        if ((lane & 15) == 0) scores[(size_t)rj[j] * Hc + head] = s;

        float t = ye1[bj[j] * DXc + col] + (ye2[bj[j] * DXc + col] + 1.f) * Av;
        T[(size_t)rj[j] * DXc + col] = t;
      }
    }
  }
}

// ---------------------------------------------------------------------------
// Softmax over k with key-mask, times em; in place. Block per (b,q,h).
// ---------------------------------------------------------------------------
__global__ void softmax_attn(float* __restrict__ scores, const int* __restrict__ nmask)
{
  int bqh = blockIdx.x;
  int h = bqh & (Hc - 1);
  int bq = bqh >> 3;
  int b_ = bq >> 8, q = bq & 255;
  int tid = threadIdx.x;
  __shared__ float red[256];

  size_t base = (size_t)bq * NNc * Hc + h;
  float s = scores[base + (size_t)tid * Hc];
  bool mk = nmask[b_ * NNc + tid] != 0;
  float ms = mk ? s : -1.0e9f;

  red[tid] = ms; __syncthreads();
  for (int st = 128; st > 0; st >>= 1) {
    if (tid < st) red[tid] = fmaxf(red[tid], red[tid + st]);
    __syncthreads();
  }
  float mx = red[0]; __syncthreads();

  float e = expf(ms - mx);
  red[tid] = e; __syncthreads();
  for (int st = 128; st > 0; st >>= 1) {
    if (tid < st) red[tid] += red[tid + st];
    __syncthreads();
  }
  float sum = red[0];

  float a = e / sum;
  if (!(a == a)) a = 0.f;
  float em = ((nmask[b_ * NNc + q] != 0) && mk) ? 1.f : 0.f;
  scores[base + (size_t)tid * Hc] = a * em;
}

__global__ void attn_v(const float* __restrict__ attn, const float* __restrict__ V,
                       float* __restrict__ wV)
{
  int idx = blockIdx.x * blockDim.x + threadIdx.x;
  if (idx >= BSc * NNc * DXc) return;
  int c = idx & 127;
  int bq = idx >> 7;
  int h = c >> 4;
  int b_ = bq >> 8;
  float acc = 0.f;
  for (int k = 0; k < NNc; ++k) {
    acc += attn[((size_t)bq * NNc + k) * Hc + h] *
           V[(size_t)(b_ * NNc + k) * DXc + c];
  }
  wV[idx] = acc;
}

__global__ void modulate_x(const float* __restrict__ wV, const float* __restrict__ yx1,
                           const float* __restrict__ yx2, float* __restrict__ out)
{
  int idx = blockIdx.x * blockDim.x + threadIdx.x;
  if (idx >= BSc * NNc * DXc) return;
  int c = idx & 127;
  int b_ = idx >> 15;
  out[idx] = yx1[b_ * DXc + c] + (yx2[b_ * DXc + c] + 1.f) * wV[idx];
}

__global__ void small_lin(const float* __restrict__ in, const float* __restrict__ W,
                          const float* __restrict__ bias, float* __restrict__ out,
                          int rows, int din, int dout, int relu)
{
  int idx = blockIdx.x * blockDim.x + threadIdx.x;
  if (idx >= rows * dout) return;
  int r = idx / dout, o = idx - r * dout;
  float acc = bias[o];
  for (int i = 0; i < din; ++i) acc += in[r * din + i] * W[o * din + i];
  if (relu) acc = fmaxf(acc, 0.f);
  out[idx] = acc;
}

__global__ void concat_y(const float* __restrict__ yp, const float* __restrict__ xq,
                         const float* __restrict__ xk, const float* __restrict__ ey,
                         float* __restrict__ out)
{
  int idx = blockIdx.x * blockDim.x + threadIdx.x;
  if (idx >= BSc * 4 * DYc) return;
  int b_ = idx / (4 * DYc), c = idx - b_ * 4 * DYc;
  float v;
  if      (c <     DYc) v = yp[b_ * DYc + c];
  else if (c < 2 * DYc) v = xq[b_ * DYc + c -     DYc];
  else if (c < 3 * DYc) v = xk[b_ * DYc + c - 2 * DYc];
  else                  v = ey[b_ * DYc + c - 3 * DYc];
  out[idx] = v;
}

__global__ void pool_e(const float* __restrict__ E, const int* __restrict__ nmask,
                       float* __restrict__ out)
{
  int b_ = blockIdx.x / DEc, c = blockIdx.x - b_ * DEc;
  int tid = threadIdx.x;
  __shared__ float sS[256], sQ[256], sN[256], sX[256], sC[256];
  float sum = 0.f, sq = 0.f, mn = 1e30f, mx = -1e30f, cnt = 0.f;
  for (int i = tid; i < NNc * NNc; i += 256) {
    int q = i >> 8, k = i & 255;
    if (nmask[b_ * NNc + q] && nmask[b_ * NNc + k]) {
      float v = E[(((size_t)b_ * NNc + q) * NNc + k) * DEc + c];
      sum += v; sq += v * v; mn = fminf(mn, v); mx = fmaxf(mx, v); cnt += 1.f;
    }
  }
  sS[tid] = sum; sQ[tid] = sq; sN[tid] = mn; sX[tid] = mx; sC[tid] = cnt;
  __syncthreads();
  for (int st = 128; st > 0; st >>= 1) {
    if (tid < st) {
      sS[tid] += sS[tid + st]; sQ[tid] += sQ[tid + st];
      sN[tid] = fminf(sN[tid], sN[tid + st]); sX[tid] = fmaxf(sX[tid], sX[tid + st]);
      sC[tid] += sC[tid + st];
    }
    __syncthreads();
  }
  if (tid == 0) {
    float denom = (sC[0] == 0.f) ? 1.f : sC[0];
    float mean = sS[0] / denom;
    float var = fmaxf(sQ[0] / denom - mean * mean, 0.f);
    out[b_ * 4 * DEc + 0 * DEc + c] = mean;
    out[b_ * 4 * DEc + 1 * DEc + c] = sqrtf(var + 1e-10f);
    out[b_ * 4 * DEc + 2 * DEc + c] = (sC[0] == 0.f) ? 0.f : sN[0];
    out[b_ * 4 * DEc + 3 * DEc + c] = (sC[0] == 0.f) ? 0.f : sX[0];
  }
}

__global__ void pool_x(const float* __restrict__ X, const int* __restrict__ nmask,
                       float* __restrict__ out)
{
  int b_ = blockIdx.x / DXc, c = blockIdx.x - b_ * DXc;
  int tid = threadIdx.x;
  __shared__ float sS[256], sQ[256], sN[256], sX[256], sC[256];
  float v = X[((size_t)b_ * NNc + tid) * DXc + c];
  bool m = nmask[b_ * NNc + tid] != 0;
  sS[tid] = m ? v : 0.f;
  sQ[tid] = m ? v * v : 0.f;
  sN[tid] = m ? v : 1e30f;
  sX[tid] = m ? v : -1e30f;
  sC[tid] = m ? 1.f : 0.f;
  __syncthreads();
  for (int st = 128; st > 0; st >>= 1) {
    if (tid < st) {
      sS[tid] += sS[tid + st]; sQ[tid] += sQ[tid + st];
      sN[tid] = fminf(sN[tid], sN[tid + st]); sX[tid] = fmaxf(sX[tid], sX[tid + st]);
      sC[tid] += sC[tid + st];
    }
    __syncthreads();
  }
  if (tid == 0) {
    float denom = (sC[0] == 0.f) ? 1.f : sC[0];
    float mean = sS[0] / denom;
    float var = fmaxf(sQ[0] / denom - mean * mean, 0.f);
    out[b_ * 4 * DXc + 0 * DXc + c] = mean;
    out[b_ * 4 * DXc + 1 * DXc + c] = sqrtf(var + 1e-10f);
    out[b_ * 4 * DXc + 2 * DXc + c] = (sC[0] == 0.f) ? 0.f : sN[0];
    out[b_ * 4 * DXc + 3 * DXc + c] = (sC[0] == 0.f) ? 0.f : sX[0];
  }
}

__global__ void ln_add(const float* __restrict__ a, const float* __restrict__ r,
                       const float* __restrict__ g, const float* __restrict__ be,
                       float* __restrict__ out, int rows, int D)
{
  int wave = (int)((blockIdx.x * blockDim.x + threadIdx.x) >> 5);
  int lane = threadIdx.x & 31;
  if (wave >= rows) return;
  const float* pa = a + (size_t)wave * D;
  const float* pr = r + (size_t)wave * D;
  float sum = 0.f, sq = 0.f;
  for (int e = lane; e < D; e += 32) {
    float v = pa[e] + pr[e];
    sum += v; sq += v * v;
  }
#pragma unroll
  for (int off = 1; off < 32; off <<= 1) {
    sum += __shfl_xor(sum, off, 32);
    sq  += __shfl_xor(sq,  off, 32);
  }
  float mu = sum / (float)D;
  float var = sq / (float)D - mu * mu;
  float inv = 1.0f / sqrtf(var + 1e-5f);
  float* po = out + (size_t)wave * D;
  for (int e = lane; e < D; e += 32) {
    float v = pa[e] + pr[e];
    po[e] = (v - mu) * inv * g[e] + be[e];
  }
}

// ---------------------------------------------------------------------------
// Host side
// ---------------------------------------------------------------------------
template<int RELU, int MASK, int TN>
static inline void launch_gemm(const float* A, const _Float16* Wh, const float* b,
                               float* out, int M, int K, int Nout, const int* nm,
                               hipStream_t s) {
  int chunks = Nout / (16 * TN);
  int waves = (M / 16) * chunks;
  gemm_wmma<RELU, MASK, TN><<<(waves + 7) / 8, 256, 0, s>>>(A, Wh, b, out, M, K, Nout, nm);
}

static inline void launch_sw(const float* W, _Float16* out, int Nout, int K, hipStream_t s) {
  int total = Nout * K;
  swizzle_w<<<(total + 255) / 256, 256, 0, s>>>(W, out, Nout, K);
}

static inline void launch_ln(const float* a, const float* r, const float* g, const float* be,
                             float* out, int rows, int D, hipStream_t s) {
  ln_add<<<(rows + 7) / 8, 256, 0, s>>>(a, r, g, be, out, rows, D);
}

static inline void launch_slin(const float* in, const float* W, const float* b, float* out,
                               int rows, int din, int dout, int relu, hipStream_t s) {
  int total = rows * dout;
  small_lin<<<(total + 255) / 256, 256, 0, s>>>(in, W, b, out, rows, din, dout, relu);
}

extern "C" void kernel_launch(void* const* d_in, const int* in_sizes, int n_in,
                              void* d_out, int out_size, void* d_ws, size_t ws_size,
                              hipStream_t stream) {
  const float* X  = (const float*)d_in[0];
  const float* E  = (const float*)d_in[1];
  const float* y  = (const float*)d_in[2];
  const int*   nm = (const int*)d_in[3];
  const float* P[64];
  for (int i = 4; i < n_in && i < 66; ++i) P[i - 4] = (const float*)d_in[i];
  // 0 qW 1 qB | 2 kW 3 kB | 4 vW 5 vB | 6 eaW 7 eaB | 8 emW 9 emB
  // 10 yeaW 11 yeaB | 12 yemW 13 yemB | 14 yxaW 15 yxaB | 16 yxmW 17 yxmB
  // 18 ypW 19 ypB | 20 rxqW 21 rxqB | 22 rxkW 23 rxkB | 24 reW 25 reB
  // 26 xoW 27 xoB | 28 eoW 29 eoB | 30 yo0W 31 yo0B | 32 yo1W 33 yo1B
  // 34 fx0W 35 fx0B | 36 fx1W 37 fx1B | 38 fe0W 39 fe0B | 40 fe1W 41 fe1B
  // 42 fy0W 43 fy0B | 44 fy1W 45 fy1B
  // 46 nX1g 47 nX1b | 48 nE1g 49 nE1b | 50 ny1g 51 ny1b
  // 52 nX3g 53 nX3b | 54 nE3g 55 nE3b | 56 ny3g 57 ny3b

  float* ws = (float*)d_ws;
  const size_t R  = (size_t)BSc * NNc * NNc;
  const size_t RX = (size_t)BSc * NNc;

  size_t oT  = 0;
  size_t oS  = oT + R * DXc;
  size_t oNE = oS + R * Hc;
  size_t oQ  = oNE + R * DEc;
  size_t oK  = oQ + RX * DXc;
  size_t oV  = oK + RX * DXc;
  size_t oWV = oV + RX * DXc;
  size_t oX1 = oWV + RX * DXc;
  size_t oX2 = oX1 + RX * DXc;
  size_t oXh = oX2 + RX * DXc;
  size_t oXf = oXh + RX * FFXc;
  size_t oSm = oXf + RX * DXc;
  float* ye1  = ws + oSm;        float* ye2  = ye1 + 512;
  float* yx1  = ye2 + 512;       float* yx2  = yx1 + 512;
  float* yp   = yx2 + 512;       float* pE   = yp + 512;
  float* pX   = pE + 1024;
  float* ey   = pX + 2048;       float* xqy  = ey + 512;
  float* xky  = xqy + 512;       float* ycat = xky + 512;
  float* yh0  = ycat + 2048;     float* ynew = yh0 + 512;
  float* yo1  = ynew + 512;      float* yffh = yo1 + 512;
  float* yff  = yffh + 2048;

  // f16 swizzled-weight region (32B aligned: all offsets multiples of 8 floats)
  size_t oWh = oSm + 32768;
  _Float16* WH = (_Float16*)(ws + oWh);
  _Float16* hQ   = WH;                 // 128x128 = 16384 halves
  _Float16* hK   = hQ   + 16384;
  _Float16* hV   = hK   + 16384;
  _Float16* hXO  = hV   + 16384;       // 128x128
  _Float16* hEO  = hXO  + 16384;       // 64x128  = 8192
  _Float16* hEA  = hEO  + 8192;        // 128x64  = 8192
  _Float16* hEM  = hEA  + 8192;
  _Float16* hFX0 = hEM  + 8192;        // 512x128 = 65536
  _Float16* hFX1 = hFX0 + 65536;       // 128x512 = 65536
  _Float16* hFE0 = hFX1 + 65536;       // 128x64  = 8192
  _Float16* hFE1 = hFE0 + 8192;        // 64x128  = 8192

  float* outX = (float*)d_out;
  float* outE = outX + RX * DXc;
  float* outY = outE + R * DEc;

  // --- weight pre-swizzle (tiny, once per call, deterministic) ---
  launch_sw(P[0],  hQ,   DXc, DXc, stream);
  launch_sw(P[2],  hK,   DXc, DXc, stream);
  launch_sw(P[4],  hV,   DXc, DXc, stream);
  launch_sw(P[26], hXO,  DXc, DXc, stream);
  launch_sw(P[28], hEO,  DEc, DXc, stream);
  launch_sw(P[6],  hEA,  DXc, DEc, stream);
  launch_sw(P[8],  hEM,  DXc, DEc, stream);
  launch_sw(P[34], hFX0, FFXc, DXc, stream);
  launch_sw(P[36], hFX1, DXc, FFXc, stream);
  launch_sw(P[38], hFE0, FFEc, DEc, stream);
  launch_sw(P[40], hFE1, DEc, FFEc, stream);

  // --- y-conditioning vectors ---
  launch_slin(y, P[10], P[11], ye1, BSc, DYc, DXc, 0, stream);
  launch_slin(y, P[12], P[13], ye2, BSc, DYc, DXc, 0, stream);
  launch_slin(y, P[14], P[15], yx1, BSc, DYc, DXc, 0, stream);
  launch_slin(y, P[16], P[17], yx2, BSc, DYc, DXc, 0, stream);
  launch_slin(y, P[18], P[19], yp,  BSc, DYc, DYc, 0, stream);

  // --- Q,K,V projections (masked rows) ---
  launch_gemm<0,1,8>(X, hQ, P[1], ws + oQ, (int)RX, DXc, DXc, nm, stream);
  launch_gemm<0,1,8>(X, hK, P[3], ws + oK, (int)RX, DXc, DXc, nm, stream);
  launch_gemm<0,1,8>(X, hV, P[5], ws + oV, (int)RX, DXc, DXc, nm, stream);

  // --- fused E1/E2 + attention modulation + scores + y-mod T ---
  {
    int waves = (int)(R / 16);
    fused_e_attn<<<(waves + 7) / 8, 256, 0, stream>>>(
        E, hEA, P[7], hEM, P[9], ws + oQ, ws + oK, ye1, ye2, nm,
        ws + oT, ws + oS);
  }

  // --- newE = e_out(T) * em ---
  launch_gemm<0,2,4>(ws + oT, hEO, P[29], ws + oNE, (int)R, DXc, DEc, nm, stream);

  // --- attention: softmax, attn @ V, x-modulation, x_out ---
  softmax_attn<<<BSc * NNc * Hc, 256, 0, stream>>>(ws + oS, nm);
  attn_v<<<(int)(RX * DXc + 255) / 256, 256, 0, stream>>>(ws + oS, ws + oV, ws + oWV);
  modulate_x<<<(int)(RX * DXc + 255) / 256, 256, 0, stream>>>(ws + oWV, yx1, yx2, ws + oX1);
  launch_gemm<0,1,8>(ws + oX1, hXO, P[27], ws + oX2, (int)RX, DXc, DXc, nm, stream);

  // --- pooled y path ---
  pool_e<<<BSc * DEc, 256, 0, stream>>>(E, nm, pE);
  pool_x<<<BSc * DXc, 256, 0, stream>>>(X, nm, pX);
  launch_slin(pE, P[24], P[25], ey,  BSc, 4 * DEc, DYc, 0, stream);
  launch_slin(pX, P[20], P[21], xqy, BSc, 4 * DXc, DYc, 0, stream);
  launch_slin(pX, P[22], P[23], xky, BSc, 4 * DXc, DYc, 0, stream);
  concat_y<<<(BSc * 4 * DYc + 255) / 256, 256, 0, stream>>>(yp, xqy, xky, ey, ycat);
  launch_slin(ycat, P[30], P[31], yh0,  BSc, 4 * DYc, DYc, 1, stream);
  launch_slin(yh0,  P[32], P[33], ynew, BSc, DYc, DYc, 0, stream);

  // --- residual + LN (block 1) ---
  launch_ln(X, ws + oX2, P[46], P[47], outX, (int)RX, DXc, stream);
  launch_ln(E, ws + oNE, P[48], P[49], outE, (int)R,  DEc, stream);
  launch_ln(y, ynew,     P[50], P[51], yo1,  BSc,     DYc, stream);

  // --- FFN X ---
  launch_gemm<1,0,8>(outX, hFX0, P[35], ws + oXh, (int)RX, DXc, FFXc, nm, stream);
  launch_gemm<0,0,8>(ws + oXh, hFX1, P[37], ws + oXf, (int)RX, FFXc, DXc, nm, stream);
  launch_ln(outX, ws + oXf, P[52], P[53], outX, (int)RX, DXc, stream);

  // --- FFN E (hidden reuses T buffer, output reuses newE buffer) ---
  launch_gemm<1,0,8>(outE, hFE0, P[39], ws + oT, (int)R, DEc, FFEc, nm, stream);
  launch_gemm<0,0,4>(ws + oT, hFE1, P[41], ws + oNE, (int)R, FFEc, DEc, nm, stream);
  launch_ln(outE, ws + oNE, P[54], P[55], outE, (int)R, DEc, stream);

  // --- FFN y ---
  launch_slin(yo1,  P[42], P[43], yffh, BSc, DYc, FFYc, 1, stream);
  launch_slin(yffh, P[44], P[45], yff,  BSc, FFYc, DYc, 0, stream);
  launch_ln(yo1, yff, P[56], P[57], outY, BSc, DYc, stream);

  (void)in_sizes; (void)out_size; (void)ws_size;
}
